// WindowCrossAttention_27444841022107
// MI455X (gfx1250) — matrix operand order repaired and checked by reading, not compiled, not run on hardware
//
#include <hip/hip_runtime.h>
#include <hip/hip_bf16.h>

typedef __bf16 bf16_t;
typedef __attribute__((ext_vector_type(16))) __bf16 v16bf;
typedef __attribute__((ext_vector_type(8)))  __bf16 v8bf;
typedef __attribute__((ext_vector_type(8)))  float   v8f;

#define DIM    512
#define NW     49
#define NWIN   2048          // B * nH = 8 * 256
#define XSTR   528           // bf16 row stride for 64x512 tiles (32B aligned rows, bank spread)
#define VTSTR  72            // bf16 row stride for 512x64 V-transposed
#define PSTR   72            // bf16 row stride for 64x64 P
#define SSTR   68            // f32 row stride for 64x64 scores
#define ATT_SCALE 0.17677669529663687f  // 32^-0.5 (HEAD_DIM^-0.5 per reference)

// ---------------------------------------------------------------------------
// Fragment loaders for V_WMMA_F32_16X16X32_BF16 (ISA 7.12.2 lane layouts).
// A (16x32 bf16): lanes 0-15 row M=lane, K = {0..7,16..23}; lanes 16-31 same
// row group, K = {8..15, 24..31}.  Two 16B contiguous loads per lane.
// ---------------------------------------------------------------------------
__device__ __forceinline__ v16bf frag_a(const bf16_t* base, int stride,
                                        int rowTile, int kChunk) {
    const int lane = threadIdx.x & 31;
    const int row  = rowTile * 16 + (lane & 15);
    const int kb   = kChunk * 32 + ((lane >> 4) << 3);   // +8 for hi half-wave
    const bf16_t* p = base + row * stride + kb;
    v8bf lo = *(const v8bf*)(p);        // K = kb .. kb+7
    v8bf hi = *(const v8bf*)(p + 16);   // K = kb+16 .. kb+23
    v16bf r;
#pragma unroll
    for (int i = 0; i < 8; ++i) { r[i] = lo[i]; r[i + 8] = hi[i]; }
    return r;
}

// B (32x16 bf16): lanes 0-15 col N=lane, K = {0..7,16..23}; lanes 16-31 col N=lane-16,
// K = {8..15,24..31}.  The storage is "row = N, contiguous K" (K stored row-major per
// output column), so this is two 16B contiguous reads per lane.
__device__ __forceinline__ v16bf frag_b(const bf16_t* base, int stride,
                                        int colTile, int kChunk) {
    const int lane = threadIdx.x & 31;
    const int n    = colTile * 16 + (lane & 15);
    const int kb   = kChunk * 32 + ((lane >> 4) << 3);   // +8 for hi half-wave
    const bf16_t* p = base + n * stride + kb;
    v8bf lo = *(const v8bf*)(p);
    v8bf hi = *(const v8bf*)(p + 16);
    v16bf r;
#pragma unroll
    for (int i = 0; i < 8; ++i) { r[i] = lo[i]; r[i + 8] = hi[i]; }
    return r;
}

// ---------------------------------------------------------------------------
// Pre-kernel: W_qkv (f32, [512][1536]) -> Wt (bf16, [1536][512]) in workspace.
// Fragment-friendly: row = output column, contiguous contraction dim.
// ---------------------------------------------------------------------------
__global__ void wqkv_transpose_bf16(const float* __restrict__ W,
                                    bf16_t* __restrict__ Wt) {
    int idx = blockIdx.x * 256 + threadIdx.x;
    if (idx < 1536 * 512) {
        int n = idx >> 9;     // output column 0..1535
        int c = idx & 511;    // contraction index
        Wt[idx] = (bf16_t)W[c * 1536 + n];
    }
}

// ---------------------------------------------------------------------------
// Fused window attention: one workgroup (8 wave32) per (b, h) window.
// ---------------------------------------------------------------------------
__global__ __launch_bounds__(256, 1)
void wca_kernel(const float* __restrict__ x,
                const bf16_t* __restrict__ Wt,
                const float* __restrict__ bqkv,
                float* __restrict__ out) {
    __shared__ __align__(32) bf16_t s_x[64 * XSTR];     // X bf16; later S (f32) + P (bf16)
    __shared__ __align__(32) bf16_t s_q[64 * XSTR];     // Q * scale, bf16
    __shared__ __align__(32) bf16_t s_k[64 * XSTR];     // K, bf16
    __shared__ __align__(32) bf16_t s_vt[512 * VTSTR];  // V transposed [c][j], bf16

    const int tid  = threadIdx.x;
    const int lane = tid & 31;
    const int wave = tid >> 5;
    const int lo16 = lane & 15;
    const int hi   = lane >> 4;
    const long base = (long)blockIdx.x * (NW * DIM);

    // ---- Phase 1: load window X -> bf16 LDS (rows 49..63 zero) --------------
    for (int i = tid; i < 64 * DIM; i += 256) {
        int row = i >> 9, col = i & 511;
        float v = (row < NW) ? x[base + row * DIM + col] : 0.0f;
        s_x[row * XSTR + col] = (bf16_t)v;
    }
    __syncthreads();

    // ---- Phase 2: QKV = X @ W + b;  96 col-tiles x 4 row-tiles -------------
    for (int nl = 0; nl < 12; ++nl) {
        const int n = wave * 12 + nl;             // output col tile 0..95
        const float bias = bqkv[n * 16 + lo16];
        v8f acc[4];
        v8f zero = {};
#pragma unroll
        for (int r = 0; r < 4; ++r) acc[r] = zero;
        for (int k = 0; k < 16; ++k) {
            v16bf bf = frag_b(Wt, DIM, n, k);
#pragma unroll
            for (int r = 0; r < 4; ++r) {
                v16bf af = frag_a(s_x, XSTR, r, k);
                acc[r] = __builtin_amdgcn_wmma_f32_16x16x32_bf16(
                    false, af, false, bf, (short)0, acc[r], false, false);
            }
        }
        const int col = n * 16 + lo16;            // global output column 0..1535
#pragma unroll
        for (int r = 0; r < 4; ++r) {
#pragma unroll
            for (int j = 0; j < 8; ++j) {
                int row = r * 16 + j + hi * 8;    // C/D layout: VGPR j -> M=j(+8)
                float v = acc[r][j] + bias;
                if (col < 512) {
                    s_q[row * XSTR + col] = (bf16_t)(v * ATT_SCALE);
                } else if (col < 1024) {
                    s_k[row * XSTR + (col - 512)] = (bf16_t)v;
                } else {
                    s_vt[(col - 1024) * VTSTR + row] = (bf16_t)v;  // transposed
                }
            }
        }
    }
    __syncthreads();

    // Overlay scores + probabilities on the (now dead) X region.
    float*  Sb = (float*)s_x;                          // 64 x SSTR f32 (17408 B)
    bf16_t* Pb = (bf16_t*)((char*)s_x + 64 * SSTR * 4); // 64 x PSTR bf16

    // ---- Phase 3: S = (Q*scale) @ K^T ; 16 tiles, 2 per wave ---------------
#pragma unroll
    for (int tl = 0; tl < 2; ++tl) {
        const int t = wave * 2 + tl;
        const int r = t >> 2, c = t & 3;
        v8f acc = {};
        for (int k = 0; k < 16; ++k) {
            v16bf af = frag_a(s_q, XSTR, r, k);
            v16bf bf = frag_b(s_k, XSTR, c, k);
            acc = __builtin_amdgcn_wmma_f32_16x16x32_bf16(
                false, af, false, bf, (short)0, acc, false, false);
        }
#pragma unroll
        for (int j = 0; j < 8; ++j) {
            int row = r * 16 + j + hi * 8;
            Sb[row * SSTR + c * 16 + lo16] = acc[j];
        }
    }
    __syncthreads();

    // ---- Phase 4: row softmax over the 49 valid columns --------------------
    if (tid < 64) {
        const int row = tid;
        if (row < NW) {
            float m = -1e30f;
            for (int j = 0; j < NW; ++j) m = fmaxf(m, Sb[row * SSTR + j]);
            float s = 0.0f;
            for (int j = 0; j < NW; ++j) s += __expf(Sb[row * SSTR + j] - m);
            const float inv = 1.0f / s;
            for (int j = 0; j < NW; ++j)
                Pb[row * PSTR + j] = (bf16_t)(__expf(Sb[row * SSTR + j] - m) * inv);
            for (int j = NW; j < 64; ++j)
                Pb[row * PSTR + j] = (bf16_t)0.0f;   // mask padded K/V rows
        } else {
            for (int j = 0; j < 64; ++j) Pb[row * PSTR + j] = (bf16_t)0.0f;
        }
    }
    __syncthreads();

    // ---- Phase 5: Out = P @ V ; 128 tiles, 16 per wave ---------------------
    for (int tl = 0; tl < 16; ++tl) {
        const int tile = wave * 16 + tl;
        const int rt = tile >> 5, ct = tile & 31;
        v8f acc = {};
#pragma unroll
        for (int k = 0; k < 2; ++k) {
            v16bf af = frag_a(Pb, PSTR, rt, k);
            v16bf bf = frag_b(s_vt, VTSTR, ct, k);   // s_vt rows = channel c
            acc = __builtin_amdgcn_wmma_f32_16x16x32_bf16(
                false, af, false, bf, (short)0, acc, false, false);
        }
#pragma unroll
        for (int j = 0; j < 8; ++j) {
            int row = rt * 16 + j + hi * 8;
            if (row < NW)
                out[base + row * DIM + ct * 16 + lo16] = acc[j];
        }
    }
}

// ---------------------------------------------------------------------------
extern "C" void kernel_launch(void* const* d_in, const int* in_sizes, int n_in,
                              void* d_out, int out_size, void* d_ws, size_t ws_size,
                              hipStream_t stream) {
    (void)in_sizes; (void)n_in; (void)out_size; (void)ws_size;
    const float* x    = (const float*)d_in[0];
    // d_in[1] (x_all) is unused by the reference.
    const float* Wq   = (const float*)d_in[2];
    const float* bq   = (const float*)d_in[3];
    bf16_t*      Wt   = (bf16_t*)d_ws;            // 1536*512 bf16 = 1.5 MB
    float*       outp = (float*)d_out;

    const int wElems = 1536 * 512;
    hipLaunchKernelGGL(wqkv_transpose_bf16, dim3((wElems + 255) / 256), dim3(256),
                       0, stream, Wq, Wt);
    hipLaunchKernelGGL(wca_kernel, dim3(NWIN), dim3(256), 0, stream,
                       x, Wt, bq, outp);
}
